// BitDiscriminatorR_41652592836877
// MI455X (gfx1250) — matrix operand order repaired and checked by reading, not compiled, not run on hardware
//
#include <hip/hip_runtime.h>
#include <hip/hip_bf16.h>
#include <cmath>

typedef __attribute__((ext_vector_type(16))) _Float16 v16h;
typedef __attribute__((ext_vector_type(8)))  float    v8f;
typedef __attribute__((ext_vector_type(4)))  unsigned int uint32x4;
typedef __attribute__((ext_vector_type(8)))  int      int32x8;
typedef __attribute__((ext_vector_type(4)))  int      int32x4;
typedef unsigned short u16;
typedef unsigned int   u32;

#define WAVES 4
#define EPSQ  1e-5f

union HFrag {
    v16h h;
    uint4 q[2];
    u16   u[16];
};

__device__ __forceinline__ u16 f2h_bits(float v) {
    union { _Float16 h; u16 u; } c;
    c.h = (_Float16)v;
    return c.u;
}

// ---------------------------------------------------------------------------
// TDM: DMA a [coutpad][kpad] f16 weight matrix from global into LDS.
// D# built per CDNA5 ISA (CDNA5 ISA): 2D tensor, data_size=2B.
// ---------------------------------------------------------------------------
__device__ __forceinline__ void tdm_load_weights(const _Float16* gsrc, u16* lds_dst,
                                                 unsigned kpad, unsigned coutpad) {
    const unsigned long long ga = (unsigned long long)(const void*)gsrc;
    const unsigned int       la = (unsigned int)(unsigned long long)(void*)lds_dst;

    uint32x4 g0 = {
        1u,                                              // count=1, user descriptor
        la,                                              // lds_addr [63:32]
        (unsigned int)(ga & 0xffffffffu),                // global_addr [95:64]
        (unsigned int)((ga >> 32) & 0x01ffffffu) | (2u << 30)  // addr hi | type=2
    };
    int32x8 g1;
    g1[0] = (int)(1u << 16);             // wg_mask=0, data_size=1 (2 bytes)
    g1[1] = (int)(kpad << 16);           // tensor_dim0[15:0] at bits[63:48]
    g1[2] = (int)(coutpad << 16);        // tensor_dim0[31:16]=0 | tensor_dim1[15:0]
    g1[3] = (int)(kpad << 16);           // tensor_dim1[31:16]=0 | tile_dim0
    g1[4] = (int)coutpad;                // tile_dim1 | tile_dim2=0
    g1[5] = (int)kpad;                   // tensor_dim0_stride[31:0]
    g1[6] = 0;                           // stride0[47:32] | stride1[15:0]
    g1[7] = 0;
    int32x4 gz = {0, 0, 0, 0};
#if defined(__clang_major__) && __clang_major__ >= 23
    int32x8 gz8 = {0, 0, 0, 0, 0, 0, 0, 0};
    __builtin_amdgcn_tensor_load_to_lds(g0, g1, gz, gz, gz8, 0);
#else
    __builtin_amdgcn_tensor_load_to_lds(g0, g1, gz, gz, 0);
#endif
}

// ---------------------------------------------------------------------------
// Reductions
// ---------------------------------------------------------------------------
__global__ void init_red_kernel(u32* red) {
    if (threadIdx.x < 16) red[threadIdx.x] = 0u;
}

__global__ void absmax_kernel(const float* __restrict__ x, int n, u32* slot) {
    __shared__ float sm[256];
    float m = 0.0f;
    for (int i = blockIdx.x * blockDim.x + threadIdx.x; i < n; i += gridDim.x * blockDim.x)
        m = fmaxf(m, fabsf(x[i]));
    sm[threadIdx.x] = m;
    __syncthreads();
    for (int s = 128; s > 0; s >>= 1) {
        if ((int)threadIdx.x < s) sm[threadIdx.x] = fmaxf(sm[threadIdx.x], sm[threadIdx.x + s]);
        __syncthreads();
    }
    if (threadIdx.x == 0) atomicMax(slot, __float_as_uint(sm[0]));  // all values >= 0
}

__global__ void abssum_kernel(const float* __restrict__ x, int n, float* slot) {
    __shared__ float sm[256];
    float a = 0.0f;
    for (int i = blockIdx.x * blockDim.x + threadIdx.x; i < n; i += gridDim.x * blockDim.x)
        a += fabsf(x[i]);
    sm[threadIdx.x] = a;
    __syncthreads();
    for (int s = 128; s > 0; s >>= 1) {
        if ((int)threadIdx.x < s) sm[threadIdx.x] += sm[threadIdx.x + s];
        __syncthreads();
    }
    if (threadIdx.x == 0) atomicAdd(slot, sm[0]);
}

// ---------------------------------------------------------------------------
// Weight packing: ternary quantize into WMMA-B-fragment-friendly [CoutPad][Kpad]
// ---------------------------------------------------------------------------
__global__ void pack_kernel(const float* __restrict__ w, int cout, int ktot, int kpad,
                            int coutpad, int cnt, const float* sumslot,
                            float* swOut, _Float16* __restrict__ bpack) {
    const float mean = fmaxf(*sumslot / (float)cnt, EPSQ);
    const float sw   = 1.0f / mean;  // w_scale: w_int = clip(round(w*sw), -1, 1)
    if (blockIdx.x == 0 && threadIdx.x == 0) *swOut = sw;
    const int total = coutpad * kpad;
    for (int i = blockIdx.x * blockDim.x + threadIdx.x; i < total; i += gridDim.x * blockDim.x) {
        const int n = i / kpad;
        const int k = i - n * kpad;
        float q = 0.0f;
        if (n < cout && k < ktot)
            q = fminf(fmaxf(rintf(w[n * ktot + k] * sw), -1.0f), 1.0f);
        bpack[i] = (_Float16)q;
    }
}

__global__ void snake_prep_kernel(const float* __restrict__ alphas, const float* __restrict__ betas,
                                  float* __restrict__ expA, float* __restrict__ invB) {
    const int i = blockIdx.x * blockDim.x + threadIdx.x;
    if (i < 160) {
        expA[i] = __expf(alphas[i]);
        invB[i] = 1.0f / (__expf(betas[i]) + 1e-9f);
    }
}

// ---------------------------------------------------------------------------
// WMMA conv: im2col GEMM, one 16-pixel M-tile per wave, Cout in 16-wide N tiles.
// Weights TDM-staged into LDS once per block; A tile staged per K-step.
// ---------------------------------------------------------------------------
struct ConvDims {
    int Win, Wout, H, P;           // P = NB * H * Wout
    const float* src;              // input activation (float, NCHW)
    float* dst;                    // fmap output (float, NCHW)
    float* dst2;                   // optional duplicate (flat head) or nullptr
    const _Float16* bpack;         // [CoutPad][Kpad] ternary f16
    const float* bias;
    const float* expA;             // per-channel exp(alpha) or nullptr
    const float* invB;             // per-channel 1/(exp(beta)+1e-9) or nullptr
    const u32* red;                // red[layer] = absmax bits of input activation
    const float* swArr;            // swArr[layer] = weight scale
    int layer;
};

template <int CIN, int KH, int KW, int SW, int PADH, int PADW,
          int KPAD, int NTILES, bool SNAKE, int COUT>
__global__ __launch_bounds__(WAVES * 32)
void bitconv_wmma_kernel(ConvDims cd) {
    constexpr int KSTEPS  = KPAD / 32;
    constexpr int KTOT    = CIN * KH * KW;
    constexpr int COUTPAD = NTILES * 16;

    __shared__ __attribute__((aligned(16))) u16 patch[WAVES][16][32];
    __shared__ __attribute__((aligned(16))) u16 bwsh[COUTPAD * KPAD];

    const int tid  = threadIdx.x;
    const int wave = tid >> 5;
    const int lane = tid & 31;
    const int half = lane >> 4;
    const int l16  = lane & 15;

    // ---- TDM: stage packed ternary weights into LDS (one DMA per block)
    if (wave == 0) {
        tdm_load_weights(cd.bpack, &bwsh[0], (unsigned)KPAD, (unsigned)COUTPAD);
        __builtin_amdgcn_s_wait_tensorcnt(0);
    }
    __syncthreads();

    const int tile  = blockIdx.x * WAVES + wave;
    const int pbase = tile << 4;
    const int HW    = cd.H * cd.Wout;

    // dequant scales
    const float amax = fmaxf(__uint_as_float(cd.red[cd.layer]), EPSQ);
    const float sa   = 127.0f / amax;                 // act scale
    const float sw   = cd.swArr[cd.layer];            // weight scale
    const float invS = 1.0f / (sa * sw);

    // this lane's im2col row (pixel) for the fill phase
    int pm = pbase + l16;
    if (pm >= cd.P) pm = cd.P - 1;
    const int nb  = pm / HW;
    const int rem = pm - nb * HW;
    const int oh  = rem / cd.Wout;
    const int ow  = rem - oh * cd.Wout;
    const int ih0 = oh - PADH;
    const int iw0 = ow * SW - PADW;

    v8f acc[NTILES];
#pragma unroll
    for (int t = 0; t < NTILES; ++t)
        acc[t] = (v8f){0.f, 0.f, 0.f, 0.f, 0.f, 0.f, 0.f, 0.f};

#pragma unroll 1
    for (int ks = 0; ks < KSTEPS; ++ks) {
        // ---- fill 16x32 quantized patch tile: lane fills row l16, cols half*16..+15
        const int colbase = half * 16;
#pragma unroll
        for (int j = 0; j < 16; ++j) {
            const int k = ks * 32 + colbase + j;
            float v = 0.0f;
            if (k < KTOT) {
                int cin, kh, kw;
                if (CIN == 1) {
                    cin = 0; kh = k / KW; kw = k - kh * KW;
                } else {
                    cin = k / (KH * KW);
                    const int r2 = k - cin * (KH * KW);
                    kh = r2 / KW; kw = r2 - kh * KW;
                }
                const int ih = ih0 + kh;
                const int iw = iw0 + kw;
                if (ih >= 0 && ih < cd.H && iw >= 0 && iw < cd.Win) {
                    const float xv = cd.src[(((size_t)nb * CIN + cin) * cd.H + ih) * cd.Win + iw];
                    v = fminf(fmaxf(rintf(xv * sa), -128.0f), 127.0f);  // int8 absmax quant
                }
            }
            patch[wave][l16][colbase + j] = f2h_bits(v);
        }

        // prefetch next K-step's activation row (global_prefetch path)
        if (ks + 1 < KSTEPS) {
            const int kn = (ks + 1) * 32 + colbase;
            if (kn < KTOT) {
                const int cinN = (CIN == 1) ? 0 : kn / (KH * KW);
                const int r2n  = kn - cinN * (KH * KW);
                const int khN  = r2n / KW;
                const int ihN  = ih0 + khN;
                if (ihN >= 0 && ihN < cd.H) {
                    const int iwN = iw0 < 0 ? 0 : iw0;
                    __builtin_prefetch(
                        &cd.src[(((size_t)nb * CIN + cinN) * cd.H + ihN) * cd.Win + iwN], 0, 3);
                }
            }
        }
        __syncthreads();

        // ---- A fragment: M=l16, K pairs split by lane-half (two b128 LDS reads)
        HFrag a;
        a.q[0] = *(const uint4*)&patch[wave][l16][half * 8];
        a.q[1] = *(const uint4*)&patch[wave][l16][16 + half * 8];

        // ---- B fragments from LDS-resident weights + WMMA
#pragma unroll
        for (int t = 0; t < NTILES; ++t) {
            const u16* bp = &bwsh[(unsigned)(t * 16 + l16) * KPAD + ks * 32 + half * 8];
            HFrag b;
            b.q[0] = *(const uint4*)(bp);
            b.q[1] = *(const uint4*)(bp + 16);
            acc[t] = __builtin_amdgcn_wmma_f32_16x16x32_f16(
                false, a.h, false, b.h, (short)0, acc[t], false, false);
        }
        __syncthreads();
    }

    // ---- epilogue: dequant + bias + snake-beta, scatter C (M=half*8+r, N=l16)
#pragma unroll
    for (int t = 0; t < NTILES; ++t) {
        const int ch = t * 16 + l16;
        if (ch >= COUT) continue;
        const float bv = cd.bias[ch];
        float ea = 0.0f, ib = 0.0f;
        if (SNAKE) { ea = cd.expA[ch]; ib = cd.invB[ch]; }
#pragma unroll
        for (int r = 0; r < 8; ++r) {
            const int m = half * 8 + r;
            const int p = pbase + m;
            if (p >= cd.P) continue;
            const int nb2  = p / HW;
            const int rem2 = p - nb2 * HW;
            const int oh2  = rem2 / cd.Wout;
            const int ow2  = rem2 - oh2 * cd.Wout;
            float y = acc[t][r] * invS + bv;
            if (SNAKE) {
                const float s = __sinf(y * ea);
                y = y + ib * s * s;
            }
            const size_t oidx = (((size_t)nb2 * COUT + ch) * cd.H + oh2) * cd.Wout + ow2;
            cd.dst[oidx] = y;
            if (cd.dst2) cd.dst2[oidx] = y;
        }
    }
}

// ---------------------------------------------------------------------------
// Host launcher
// ---------------------------------------------------------------------------
static inline int imin(int a, int b) { return a < b ? a : b; }

extern "C" void kernel_launch(void* const* d_in, const int* in_sizes, int n_in,
                              void* d_out, int out_size, void* d_ws, size_t ws_size,
                              hipStream_t stream) {
    (void)in_sizes; (void)n_in; (void)out_size; (void)ws_size;

    const float* x = (const float*)d_in[0];
    const float* w[6] = { (const float*)d_in[1], (const float*)d_in[3], (const float*)d_in[5],
                          (const float*)d_in[7], (const float*)d_in[9], (const float*)d_in[11] };
    const float* b[6] = { (const float*)d_in[2], (const float*)d_in[4], (const float*)d_in[6],
                          (const float*)d_in[8], (const float*)d_in[10], (const float*)d_in[12] };
    const float* alphas = (const float*)d_in[13];
    const float* betas  = (const float*)d_in[14];
    float* out = (float*)d_out;

    // ---- geometry
    const int NB = 8, H = 513;
    const int Wio[7] = {600, 600, 300, 150, 75, 75, 75};  // width entering layer i; out = Wio[i+1]
    const int Cout[6] = {32, 32, 32, 32, 32, 1};

    // d_out layout: flat head first, then fmap0..fmap5
    const size_t flatN = (size_t)NB * 513 * 75;  // == fmap5 size, identical layout
    size_t fsz[6], foff[6];
    size_t o = flatN;
    for (int i = 0; i < 6; ++i) {
        fsz[i]  = (size_t)NB * Cout[i] * H * Wio[i + 1];
        foff[i] = o;
        o += fsz[i];
    }

    // ---- workspace layout
    unsigned char* ws = (unsigned char*)d_ws;
    u32*   red   = (u32*)ws;                       // [0..5] act absmax bits, [6..11] w abs-sum (float)
    float* swArr = (float*)(ws + 64);              // 6 floats
    float* expA  = (float*)(ws + 128);             // 160 floats
    float* invB  = (float*)(ws + 768);             // 160 floats
    _Float16* bp[6];
    bp[0] = (_Float16*)(ws + 1408);                // 32*32
    bp[1] = bp[0] + 32 * 32;                       // 32*864
    bp[2] = bp[1] + 32 * 864;
    bp[3] = bp[2] + 32 * 864;
    bp[4] = bp[3] + 32 * 864;                      // 32*288
    bp[5] = bp[4] + 32 * 288;                      // 16*288

    // ---- 1) zero reduction slots
    init_red_kernel<<<1, 64, 0, stream>>>(red);

    // ---- 2) weight abs-sums, then ternary pack (+scales)
    const int wcnt[6]    = {32 * 27, 32 * 864, 32 * 864, 32 * 864, 32 * 288, 1 * 288};
    const int ktot[6]    = {27, 864, 864, 864, 288, 288};
    const int kpad[6]    = {32, 864, 864, 864, 288, 288};
    const int coutpad[6] = {32, 32, 32, 32, 32, 16};
    for (int i = 0; i < 6; ++i) {
        const int blk = imin((wcnt[i] + 255) / 256, 64);
        abssum_kernel<<<blk, 256, 0, stream>>>(w[i], wcnt[i], (float*)red + 6 + i);
    }
    for (int i = 0; i < 6; ++i) {
        const int total = coutpad[i] * kpad[i];
        pack_kernel<<<(total + 255) / 256, 256, 0, stream>>>(
            w[i], Cout[i], ktot[i], kpad[i], coutpad[i], wcnt[i],
            (const float*)red + 6 + i, swArr + i, bp[i]);
    }
    snake_prep_kernel<<<1, 256, 0, stream>>>(alphas, betas, expA, invB);

    // ---- 3) per-layer: activation absmax, then WMMA conv
    const float* srcs[6] = {x, out + foff[0], out + foff[1], out + foff[2],
                            out + foff[3], out + foff[4]};
    const int srcCin[6] = {1, 32, 32, 32, 32, 32};

    for (int i = 0; i < 6; ++i) {
        const int n = NB * srcCin[i] * H * Wio[i];
        const int blk = imin((n + 255) / 256, 4096);
        absmax_kernel<<<blk, 256, 0, stream>>>(srcs[i], n, red + i);

        ConvDims cd;
        cd.Win   = Wio[i];
        cd.Wout  = Wio[i + 1];
        cd.H     = H;
        cd.P     = NB * H * Wio[i + 1];
        cd.src   = srcs[i];
        cd.dst   = out + foff[i];
        cd.dst2  = (i == 5) ? out : nullptr;  // flat head duplicates fmap5
        cd.bpack = bp[i];
        cd.bias  = b[i];
        cd.expA  = (i < 5) ? expA + 32 * i : nullptr;
        cd.invB  = (i < 5) ? invB + 32 * i : nullptr;
        cd.red   = red;
        cd.swArr = swArr;
        cd.layer = i;

        const int tiles  = (cd.P + 15) >> 4;
        const int blocks = (tiles + WAVES - 1) / WAVES;
        switch (i) {
        case 0: bitconv_wmma_kernel<1, 3, 9, 1, 1, 4,  32, 2, true, 32>
                    <<<blocks, WAVES * 32, 0, stream>>>(cd); break;
        case 1: bitconv_wmma_kernel<32, 3, 9, 2, 1, 4, 864, 2, true, 32>
                    <<<blocks, WAVES * 32, 0, stream>>>(cd); break;
        case 2: bitconv_wmma_kernel<32, 3, 9, 2, 1, 4, 864, 2, true, 32>
                    <<<blocks, WAVES * 32, 0, stream>>>(cd); break;
        case 3: bitconv_wmma_kernel<32, 3, 9, 2, 1, 4, 864, 2, true, 32>
                    <<<blocks, WAVES * 32, 0, stream>>>(cd); break;
        case 4: bitconv_wmma_kernel<32, 3, 3, 1, 1, 1, 288, 2, true, 32>
                    <<<blocks, WAVES * 32, 0, stream>>>(cd); break;
        case 5: bitconv_wmma_kernel<32, 3, 3, 1, 1, 1, 288, 1, false, 1>
                    <<<blocks, WAVES * 32, 0, stream>>>(cd); break;
        }
    }
}
